// Bilinear_Layer_55241869361591
// MI455X (gfx1250) — matrix operand order repaired and checked by reading, not compiled, not run on hardware
//
#include <hip/hip_runtime.h>
#include <hip/hip_bf16.h>
#include <stdint.h>

// Bilinear per-type score: out[n] = sigmoid(u[n]^T W[type[n]] v[n])
// N=131072, D=256, K=8.
//
// Strategy: bucket samples by type (counting sort in ws), then per-type
// GEMM tiles with v_wmma_f32_16x16x32_bf16 (f32 accumulate). Memory-bound
// (~258 MB @ 23.3 TB/s ~= 11us); bf16 WMMA compute ~7us, so WMMA keeps us
// off the compute roof that fp32 VALU math would hit.
//
// ws layout: [0,1MB) Wt bf16 transposed (k,e,d); [1MB,+128B) control ints
// (counts[8] @0, offsets[9] @8, cursors[8] @17); order[N] ints after.
// Required ws = 1MB + 128 + 4*N ~= 1.5 MB.

#define D 256
#define KTYPES 8

typedef __attribute__((ext_vector_type(16))) __bf16 v16bf;
typedef __attribute__((ext_vector_type(8)))  float  v8f;

union BF16x16 { v16bf v; unsigned short s[16]; uint4 q[2]; };

__device__ __forceinline__ unsigned short f2bf(float f) {
  unsigned int u = __float_as_uint(f);
  unsigned int r = u + 0x7FFFu + ((u >> 16) & 1u);   // round-to-nearest-even
  return (unsigned short)(r >> 16);
}

__device__ __forceinline__ void pack8(unsigned short* dst, float4 a, float4 b) {
  dst[0] = f2bf(a.x); dst[1] = f2bf(a.y); dst[2] = f2bf(a.z); dst[3] = f2bf(a.w);
  dst[4] = f2bf(b.x); dst[5] = f2bf(b.y); dst[6] = f2bf(b.z); dst[7] = f2bf(b.w);
}

__global__ void k_init(int* ctrl) {
  if (threadIdx.x < 32) ctrl[threadIdx.x] = 0;
}

__global__ void k_count(const int* __restrict__ t, int* __restrict__ counts, int n) {
  int i = blockIdx.x * blockDim.x + threadIdx.x;
  if (i < n) atomicAdd(&counts[t[i]], 1);
}

__global__ void k_scan(int* ctrl) {
  if (threadIdx.x == 0) {
    int a = 0;
    for (int i = 0; i < KTYPES; i++) { ctrl[8 + i] = a; a += ctrl[i]; }
    ctrl[8 + KTYPES] = a;
  }
}

__global__ void k_scatter(const int* __restrict__ t, int* ctrl,
                          int* __restrict__ order, int n) {
  int i = blockIdx.x * blockDim.x + threadIdx.x;
  if (i < n) {
    int k = t[i];
    int p = atomicAdd(&ctrl[17 + k], 1);
    order[ctrl[8 + k] + p] = i;
  }
}

// W[k][d][e] fp32 -> Wt[k][e][d] bf16 (so WMMA B lanes read contiguous d)
__global__ void k_convert(const float* __restrict__ W, unsigned short* __restrict__ Wt) {
  int i = blockIdx.x * blockDim.x + threadIdx.x;   // i < K*D*D
  int rem = i & (D * D - 1);
  int d = rem >> 8, e = rem & (D - 1);
  Wt[(i & ~(D * D - 1)) + (e << 8) + d] = f2bf(W[i]);
}

__global__ __launch_bounds__(32) void k_bilinear(
    const float* __restrict__ u, const float* __restrict__ v,
    const unsigned short* __restrict__ Wt, const int* __restrict__ ctrl,
    const int* __restrict__ order, float* __restrict__ out) {
  __shared__ float sU[16 * D];
  __shared__ float sV[16 * D];
  __shared__ float sRed[16];

  const int lane = threadIdx.x;
  const int m    = lane & 15;    // row within tile / column of C
  const int half = lane >> 4;

  // Bucket offsets (uniform across wave).
  int offs[KTYPES + 1];
#pragma unroll
  for (int i = 0; i <= KTYPES; i++) offs[i] = ctrl[8 + i];

  int tile = blockIdx.x;
  int k = -1, bt = 0, cnt = 0, tstart = 0;
#pragma unroll
  for (int i = 0; i < KTYPES; i++) {
    int c = offs[i + 1] - offs[i];
    int nt = (c + 15) >> 4;
    if (k < 0 && tile < tstart + nt) { k = i; bt = tile - tstart; cnt = c; }
    tstart += nt;
  }
  if (k < 0) return;   // uniform: whole wave exits, EXEC stays full otherwise

  const int rib  = bt * 16 + m;                       // row in bucket
  const int ridx = order[offs[k] + (rib < cnt ? rib : 0)];  // clamp padding rows

  // Stage 16 rows of u and v into LDS (2 lanes per row, 128 floats each).
  {
    const float4* pu = (const float4*)(u + (size_t)ridx * D + half * 128);
    const float4* pv = (const float4*)(v + (size_t)ridx * D + half * 128);
    float4* du = (float4*)(sU + m * D + half * 128);
    float4* dv = (float4*)(sV + m * D + half * 128);
#pragma unroll
    for (int t = 0; t < 32; t++) { du[t] = pu[t]; dv[t] = pv[t]; }
  }
  __syncthreads();

  // Resident A fragments: 16-bit A 16x32 layout — lanes 0-15 hold K
  // {0-7,16-23}, lanes 16-31 hold {8-15,24-31} of each 32-wide slab.
  BF16x16 afr[8];
#pragma unroll
  for (int kk = 0; kk < 8; kk++) {
    const float* base = sU + m * D + kk * 32 + half * 8;
    const float4* p0 = (const float4*)(base);
    const float4* p1 = (const float4*)(base + 16);
    pack8(&afr[kk].s[0], p0[0], p0[1]);
    pack8(&afr[kk].s[8], p1[0], p1[1]);
  }

  float acc[8];
#pragma unroll
  for (int r = 0; r < 8; r++) acc[r] = 0.0f;

  const unsigned short* Wk = Wt + ((size_t)k << 16);

#pragma unroll 1
  for (int j = 0; j < 16; j++) {       // output-column tiles of T = U*Wk
    const int e = j * 16 + m;
    const uint4* pb = (const uint4*)(Wk + e * D);   // Wt[k][e][*], contiguous d
    v8f c = {};
#pragma unroll
    for (int kk = 0; kk < 8; kk++) {
      BF16x16 b;                        // B 32x16: lanes 0-15 K={0-7,16-23}, 16-31 K={8-15,24-31}
      b.q[0] = pb[kk * 4 + half];
      b.q[1] = pb[kk * 4 + 2 + half];
      c = __builtin_amdgcn_wmma_f32_16x16x32_bf16(false, afr[kk].v, false, b.v,
                                                  (short)0, c, false, false);
    }
    // C layout: lane n=e-col, VGPR r -> row r + half*8. Fuse dot with v tile.
#pragma unroll
    for (int r = 0; r < 8; r++)
      acc[r] += c[r] * sV[(r + half * 8) * D + e];
  }

  // Reduce over the 16 e-lanes of each half (xor stays within the half).
#pragma unroll
  for (int r = 0; r < 8; r++) {
    float x = acc[r];
    x += __shfl_xor(x, 1, 32);
    x += __shfl_xor(x, 2, 32);
    x += __shfl_xor(x, 4, 32);
    x += __shfl_xor(x, 8, 32);
    acc[r] = x;
  }
  if (m == 0) {                         // lanes 0 (rows 0-7) and 16 (rows 8-15)
#pragma unroll
    for (int r = 0; r < 8; r++) sRed[half * 8 + r] = acc[r];
  }
  __syncthreads();

  if (lane < 16) {
    int row = bt * 16 + lane;
    if (row < cnt) {
      int rid = order[offs[k] + row];
      float x = sRed[lane];
      out[rid] = 1.0f / (1.0f + __expf(-x));
    }
  }
}

extern "C" void kernel_launch(void* const* d_in, const int* in_sizes, int n_in,
                              void* d_out, int out_size, void* d_ws, size_t ws_size,
                              hipStream_t stream) {
  const float* u  = (const float*)d_in[0];
  const float* v  = (const float*)d_in[1];
  const float* W  = (const float*)d_in[2];
  const int*   ti = (const int*)d_in[3];
  float* out = (float*)d_out;
  const int n = in_sizes[3];

  unsigned short* Wt = (unsigned short*)d_ws;               // 1 MB bf16 weights
  int* ctrl  = (int*)((char*)d_ws + (1 << 20));             // 32 control ints
  int* order = ctrl + 32;                                   // n sample indices

  k_init   <<<1, 32, 0, stream>>>(ctrl);
  k_count  <<<(n + 255) / 256, 256, 0, stream>>>(ti, ctrl, n);
  k_scan   <<<1, 1, 0, stream>>>(ctrl);
  k_scatter<<<(n + 255) / 256, 256, 0, stream>>>(ti, ctrl, order, n);
  k_convert<<<(KTYPES * D * D) / 256, 256, 0, stream>>>(W, Wt);

  const int tiles = (n + 15) / 16 + KTYPES;   // worst-case padded tile count
  k_bilinear<<<tiles, 32, 0, stream>>>(u, v, Wt, ctrl, order, out);
}
